// GRU_1649267442125
// MI455X (gfx1250) — matrix operand order repaired and checked by reading, not compiled, not run on hardware
//
#include <hip/hip_runtime.h>

// ---------------------------------------------------------------------------
// GRU (B=64, T=2048, I=H=128) for MI455X / gfx1250, wave32 + WMMA bf16.
//
// Phase 1 (parallel): gi = x @ w_ih^T + b_ih  -> d_ws  [B*T, 384] f32 (192 MB,
//          fits global L2). v_wmma_f32_16x16x32_bf16, f32 accumulate.
// Phase 2 (serial in T, parallel in batch): 4 workgroups x 16 batch rows.
//          w_hh lives in LDS as bf16 (~96 KB; needs CDNA5's 320 KB LDS/WGP),
//          h and gh tiles in LDS, 96 WMMAs per WG per timestep.
// ---------------------------------------------------------------------------

typedef __attribute__((ext_vector_type(16))) __bf16 v16bf;
typedef __attribute__((ext_vector_type(8)))  float  v8f;

#define BB 64
#define TT 2048
#define II 128
#define HH 128
#define GG 384            // 3*H
#define WSTR 136          // padded bf16 row stride for w in LDS (breaks bank conflicts)
#define HSTR 132          // padded f32 row stride for h in LDS

__device__ __forceinline__ float sigmoidf_(float v) {
    return 1.0f / (1.0f + __expf(-v));
}

// ---------------------------------------------------------------------------
// Kernel 1: gi[row, n] = sum_k x[row,k] * w_ih[n,k] + b_ih[n]
// grid = 1024 blocks * 8 waves -> 8192 M-tiles of 16 rows. Each wave keeps its
// A fragments (one x row per lane) in VGPRs and sweeps all 24 N-tiles reading
// bf16 B fragments from LDS.
// ---------------------------------------------------------------------------
__global__ __launch_bounds__(256) void gru_gi_kernel(
    const float* __restrict__ x,      // [B*T, I]
    const float* __restrict__ w_ih,   // [G, I]
    const float* __restrict__ b_ih,   // [G]
    float* __restrict__ gi)           // [B*T, G]
{
    extern __shared__ char smem[];
    __bf16* wl = (__bf16*)smem;       // [G][WSTR] bf16

    // Cooperative load + f32->bf16 convert of w_ih into LDS.
    for (int idx = threadIdx.x; idx < GG * II; idx += 256) {
        int n = idx / II, k = idx % II;
        wl[n * WSTR + k] = (__bf16)w_ih[idx];
    }
    __syncthreads();

    const int wave   = threadIdx.x >> 5;
    const int lane   = threadIdx.x & 31;
    const int laneHi = lane >> 4;     // 0: lanes 0-15, 1: lanes 16-31
    const int lane16 = lane & 15;
    const int mtile  = blockIdx.x * 8 + wave;
    const long row   = (long)mtile * 16 + lane16;

    // A fragments: 16-bit A 16x32 layout — lane holds row M=lane16,
    // K runs [kt*32 + laneHi*8, +8) and [kt*32 + 16 + laneHi*8, +8).
    v16bf a[4];
    const float* xr = x + row * II;
    #pragma unroll
    for (int kt = 0; kt < 4; ++kt) {
        const int k1 = kt * 32 + laneHi * 8;
        #pragma unroll
        for (int j = 0; j < 8; ++j) a[kt][j]     = (__bf16)xr[k1 + j];
        #pragma unroll
        for (int j = 0; j < 8; ++j) a[kt][8 + j] = (__bf16)xr[k1 + 16 + j];
    }

    for (int nt = 0; nt < 24; ++nt) {
        const int n = nt * 16 + lane16;
        const float bias = b_ih[n];
        v8f acc;
        #pragma unroll
        for (int r = 0; r < 8; ++r) acc[r] = bias;

        #pragma unroll
        for (int kt = 0; kt < 4; ++kt) {
            // B fragment: lane = col n, K = kt*32 + laneHi*8 + (j&7) + (j>>3)*16.
            const __bf16* wrow = wl + n * WSTR + kt * 32 + laneHi * 8;
            v16bf bf;
            #pragma unroll
            for (int j = 0; j < 16; ++j) bf[j] = wrow[(j & 7) + (j >> 3) * 16];
            acc = __builtin_amdgcn_wmma_f32_16x16x32_bf16(
                false, a[kt], false, bf, (short)0, acc, false, false);
        }
        // D layout: VGPR r -> row (laneHi*8 + r), col n.
        float* o = gi + ((long)mtile * 16 + laneHi * 8) * GG + n;
        #pragma unroll
        for (int r = 0; r < 8; ++r) o[(long)r * GG] = acc[r];
    }
}

// ---------------------------------------------------------------------------
// Kernel 2: the recurrence. One workgroup owns 16 batch rows for all T steps.
// Per step: gh = h @ w_hh^T + b_hh via WMMA (8 waves x 3 N-tiles x 4 K-tiles),
// then fused gate math + variable-length masking, h kept in LDS.
// ---------------------------------------------------------------------------
__global__ __launch_bounds__(256) void gru_rec_kernel(
    const float* __restrict__ gi,          // [B*T, G]
    const float* __restrict__ w_hh,        // [G, H]
    const float* __restrict__ b_hh,        // [G]
    const long long* __restrict__ seq_len, // [B] int64
    float* __restrict__ out)               // [B, T, H]
{
    extern __shared__ char smem[];
    __bf16* wl  = (__bf16*)smem;                              // [G][WSTR] bf16
    float*  hl  = (float*)(smem + GG * WSTR * 2);             // [16][HSTR] f32
    float*  ghl = (float*)(smem + GG * WSTR * 2 + 16 * HSTR * 4); // [16][G] f32

    for (int idx = threadIdx.x; idx < GG * HH; idx += 256) {
        int n = idx / HH, k = idx % HH;
        wl[n * WSTR + k] = (__bf16)w_hh[idx];
    }
    for (int idx = threadIdx.x; idx < 16 * HSTR; idx += 256) hl[idx] = 0.0f;
    __syncthreads();

    const int wave   = threadIdx.x >> 5;
    const int lane   = threadIdx.x & 31;
    const int laneHi = lane >> 4;
    const int lane16 = lane & 15;
    const int bbase  = blockIdx.x * 16;     // this WG's first batch row

    // Per-wave bias values for its 3 N-tiles (constant over t).
    float bias3[3];
    #pragma unroll
    for (int i = 0; i < 3; ++i) bias3[i] = b_hh[(wave * 3 + i) * 16 + lane16];

    // Elementwise mapping: element e = tid + 256*i -> m = tid/128 + 2*i.
    long long sl[8];
    #pragma unroll
    for (int i = 0; i < 8; ++i) sl[i] = seq_len[bbase + (threadIdx.x >> 7) + 2 * i];

    for (int t = 0; t < TT; ++t) {
        // ---- A fragments from h in LDS (f32 -> bf16) ----
        v16bf a[4];
        const float* hr = hl + lane16 * HSTR;
        #pragma unroll
        for (int kt = 0; kt < 4; ++kt) {
            const int k1 = kt * 32 + laneHi * 8;
            #pragma unroll
            for (int j = 0; j < 8; ++j) a[kt][j]     = (__bf16)hr[k1 + j];
            #pragma unroll
            for (int j = 0; j < 8; ++j) a[kt][8 + j] = (__bf16)hr[k1 + 16 + j];
        }

        // ---- 3 N-tiles per wave, 4 chained WMMAs each ----
        #pragma unroll
        for (int i = 0; i < 3; ++i) {
            const int nt = wave * 3 + i;
            const int n  = nt * 16 + lane16;
            v8f acc;
            #pragma unroll
            for (int r = 0; r < 8; ++r) acc[r] = bias3[i];
            #pragma unroll
            for (int kt = 0; kt < 4; ++kt) {
                const __bf16* wrow = wl + n * WSTR + kt * 32 + laneHi * 8;
                v16bf bf;
                #pragma unroll
                for (int j = 0; j < 16; ++j) bf[j] = wrow[(j & 7) + (j >> 3) * 16];
                acc = __builtin_amdgcn_wmma_f32_16x16x32_bf16(
                    false, a[kt], false, bf, (short)0, acc, false, false);
            }
            #pragma unroll
            for (int r = 0; r < 8; ++r)
                ghl[(laneHi * 8 + r) * GG + n] = acc[r];
        }
        __syncthreads();

        // ---- Fused gates + masking; each thread owns 8 (m,c) elements ----
        #pragma unroll
        for (int i = 0; i < 8; ++i) {
            const int e = threadIdx.x + 256 * i;
            const int m = e >> 7;           // row within the 16-batch tile
            const int c = e & 127;          // hidden unit
            const int b = bbase + m;
            const float* gir = gi + ((long)b * TT + t) * GG;

            const float hprev = hl[m * HSTR + c];
            const float r  = sigmoidf_(gir[c]          + ghl[m * GG + c]);
            const float z  = sigmoidf_(gir[HH + c]     + ghl[m * GG + HH + c]);
            const float nn = tanhf(gir[2 * HH + c] + r * ghl[m * GG + 2 * HH + c]);
            const float hnew = (1.0f - z) * nn + z * hprev;
            const float hnext = ((long long)t < sl[i]) ? hnew : hprev;

            hl[m * HSTR + c] = hnext;
            out[((long)b * TT + t) * HH + c] = hnext;

            // Prefetch next timestep's gi row (global_prefetch_b8); gi mostly
            // lives in the 192 MB L2 but this hides any HBM miss latency.
            if (t + 1 < TT) {
                __builtin_prefetch(gir + GG + c, 0, 0);
                __builtin_prefetch(gir + GG + HH + c, 0, 0);
                __builtin_prefetch(gir + GG + 2 * HH + c, 0, 0);
            }
        }
        __syncthreads();
    }
}

// ---------------------------------------------------------------------------
extern "C" void kernel_launch(void* const* d_in, const int* in_sizes, int n_in,
                              void* d_out, int out_size, void* d_ws, size_t ws_size,
                              hipStream_t stream) {
    const float*     x       = (const float*)d_in[0];      // [B,T,I] f32
    const long long* seq_len = (const long long*)d_in[1];  // [B] int64
    const float*     w_ih    = (const float*)d_in[2];      // [3H,I]
    const float*     w_hh    = (const float*)d_in[3];      // [3H,H]
    const float*     b_ih    = (const float*)d_in[4];      // [3H]
    const float*     b_hh    = (const float*)d_in[5];      // [3H]
    float*           out     = (float*)d_out;              // [B,T,H]
    float*           gi      = (float*)d_ws;               // [B*T,3H] = 192 MB scratch

    const size_t smem1 = (size_t)GG * WSTR * 2;                              // 104448 B
    const size_t smem2 = (size_t)GG * WSTR * 2 + 16 * HSTR * 4 + 16 * GG * 4; // 137472 B

    // Phase 1: 1024 blocks x 8 waves -> 8192 M-tiles of 16 rows.
    gru_gi_kernel<<<(BB * TT) / (16 * 8), 256, smem1, stream>>>(x, w_ih, b_ih, gi);
    // Phase 2: 4 workgroups, 16 batch rows each, serial over T.
    gru_rec_kernel<<<BB / 16, 256, smem2, stream>>>(gi, w_hh, b_hh, seq_len, out);
}
